// SingleWindowAttention_1116691497799
// MI455X (gfx1250) — matrix operand order repaired and checked by reading, not compiled, not run on hardware
//
#include <hip/hip_runtime.h>
#include <hip/hip_bf16.h>
#include <stdint.h>

typedef __bf16 v16bf __attribute__((ext_vector_type(16)));
typedef __bf16 v8bf  __attribute__((ext_vector_type(8)));
typedef float  v8f   __attribute__((ext_vector_type(8)));

__device__ __forceinline__ unsigned short f2bf(float f) {
  union { __bf16 b; unsigned short u; } cv;
  cv.b = (__bf16)f;                 // native fptrunc -> hw cvt on gfx1250
  return cv.u;
}

// Build a 16-element bf16 fragment from two aligned 16-byte chunks.
__device__ __forceinline__ v16bf frag2(const unsigned short* c0, const unsigned short* c1) {
  union { v16bf v; v8bf h[2]; } u;
  u.h[0] = *(const v8bf*)c0;
  u.h[1] = *(const v8bf*)c1;
  return u.v;
}

// ---------------- prep: transpose/convert weights to bf16, expand bias ----------------
// ws layout: [0)        qw_t  bf16 [384][128]   98304 B
//            [98304)    pw_t  bf16 [128][128]   32768 B
//            [131072)   bias  f32  [4][64][64]  65536 B
__global__ __launch_bounds__(256) void swin_prep_kernel(
    const float* __restrict__ qkv_w,       // [128][384]
    const float* __restrict__ proj_w,      // [128][128]
    const float* __restrict__ bias_table,  // [225][4]
    unsigned short* __restrict__ qw_t,
    unsigned short* __restrict__ pw_t,
    float* __restrict__ bias)
{
  int gid = blockIdx.x * 256 + threadIdx.x;
  if (gid < 49152) {
    int f = gid >> 7, c = gid & 127;
    qw_t[gid] = f2bf(qkv_w[c * 384 + f]);
  } else if (gid < 65536) {
    int e = gid - 49152;
    int f = e >> 7, c = e & 127;
    pw_t[e] = f2bf(proj_w[c * 128 + f]);
  } else if (gid < 81920) {
    int e = gid - 65536;
    int h = e >> 12, r = e & 4095;
    int q = r >> 6, kk = r & 63;
    int qi = q >> 3, qj = q & 7, ki = kk >> 3, kj = kk & 7;
    int idx = (qi - ki + 7) * 15 + (qj - kj + 7);   // (2*8-1)=15 grid
    bias[e] = bias_table[idx * 4 + h];
  }
}

// ---------------- main: fused window attention, one block per window ----------------
#define XSTRIDE 136   // 128 + 8 bf16 pad -> 272B row stride, 16B aligned, bank-friendly

__global__ __launch_bounds__(256) void swin_main_kernel(
    const float* __restrict__ x,           // [4096][64][128]
    const float* __restrict__ qkv_b,       // [384]
    const float* __restrict__ proj_b,      // [128]
    const unsigned short* __restrict__ qw_t,  // bf16 [384][128] (transposed)
    const unsigned short* __restrict__ pw_t,  // bf16 [128][128] (transposed)
    const float* __restrict__ bias,        // [4][64][64]
    float* __restrict__ out)               // [4096][64][128]
{
  // LDS: scores f32 [4][64][64] (64K, overlaid by bf16 P in place)
  //      q/k bf16 [4][64][32] (16K each), v_t bf16 [4][32][64] (16K)
  //      x/attn_out bf16 [64][136] (17408B) -- total 132096 B
  __shared__ __align__(16) unsigned char smem[132096];
  float*          sc = (float*)(smem);                     // scores / P
  unsigned short* qs = (unsigned short*)(smem + 65536);
  unsigned short* ks = (unsigned short*)(smem + 81920);
  unsigned short* vs = (unsigned short*)(smem + 98304);
  unsigned short* xs = (unsigned short*)(smem + 114688);

  const int tid  = threadIdx.x;
  const int lane = tid & 31;
  const int w    = tid >> 5;
  const int lm   = lane & 15;
  const int lh   = lane >> 4;
  const int blk  = blockIdx.x;

  // ---- phase 0: load x (float4), convert, packed 8B LDS stores ----
  const float* xg = x + (size_t)blk * 8192;
  #pragma unroll
  for (int i = 0; i < 8; ++i) {
    int e4 = tid + i * 256;             // float4 index, 2048 total
    int r = e4 >> 5, c4 = e4 & 31;
    float4 f = ((const float4*)xg)[e4];
    union { unsigned short h[4]; uint2 u; } pk;
    pk.h[0] = f2bf(f.x); pk.h[1] = f2bf(f.y);
    pk.h[2] = f2bf(f.z); pk.h[3] = f2bf(f.w);
    *(uint2*)&xs[r * XSTRIDE + c4 * 4] = pk.u;
  }
  __syncthreads();

  // ---- phase 1: QKV = x @ Wqkv  (96 tiles, K=128 -> 4 wmma steps) ----
  // ct = w + 8*s  =>  s is a compile-time unroll index: branch-free q/k/v scatter.
  const float scale = 0.17677669529663687f;  // 32^-0.5
  const int h1  = w >> 1;        // head for this wave's column group
  const int dh1 = w & 1;         // d-half
  #pragma unroll
  for (int s = 0; s < 3; ++s) {
    const int ct  = w + 8 * s;
    const int col = ct * 16 + lm;
    const unsigned short* bcol = qw_t + (size_t)col * 128;
    const float bv = qkv_b[col];
    for (int mt = 0; mt < 4; ++mt) {
      const unsigned short* arow = xs + (mt * 16 + lm) * XSTRIDE;
      v8f acc = {};
      #pragma unroll
      for (int t = 0; t < 4; ++t) {
        v16bf a  = frag2(arow + t * 32 + lh * 8, arow + t * 32 + 16 + lh * 8);
        v16bf bb = frag2(bcol + t * 32 + lh * 8, bcol + t * 32 + 16 + lh * 8);
        acc = __builtin_amdgcn_wmma_f32_16x16x32_bf16(false, a, false, bb, (short)0, acc, false, false);
      }
      #pragma unroll
      for (int r = 0; r < 8; ++r) {
        float vv = acc[r] + bv;
        int token = mt * 16 + r + 8 * lh;
        if (s == 0)      qs[(h1 * 64 + token) * 32 + dh1 * 16 + lm] = f2bf(vv * scale);
        else if (s == 1) ks[(h1 * 64 + token) * 32 + dh1 * 16 + lm] = f2bf(vv);
        else             vs[(h1 * 32 + dh1 * 16 + lm) * 64 + token] = f2bf(vv);  // V^T
      }
    }
  }
  __syncthreads();

  // ---- phase 2: scores = Q @ K^T + bias  (64 tiles, K=D=32 -> ONE wmma each) ----
  #pragma unroll 2
  for (int i = 0; i < 8; ++i) {
    int tt = w * 8 + i;
    int h = tt >> 4, qt = (tt >> 2) & 3, kt = tt & 3;
    const unsigned short* arow = qs + (h * 64 + qt * 16 + lm) * 32;
    const unsigned short* bcol = ks + (h * 64 + kt * 16 + lm) * 32;
    v16bf a  = frag2(arow + lh * 8, arow + 16 + lh * 8);
    v16bf bb = frag2(bcol + lh * 8, bcol + 16 + lh * 8);
    v8f acc = {};
    acc = __builtin_amdgcn_wmma_f32_16x16x32_bf16(false, a, false, bb, (short)0, acc, false, false);
    const float* bptr = bias + h * 4096 + kt * 16 + lm;
    #pragma unroll
    for (int r = 0; r < 8; ++r) {
      int q = qt * 16 + r + 8 * lh;
      sc[(h * 64 + q) * 64 + kt * 16 + lm] = acc[r] + bptr[q * 64];
    }
  }
  __syncthreads();

  // ---- phase 2b: softmax, one thread per (h,q) row; write bf16 P in place ----
  {
    const float* rp = sc + tid * 64;
    float vreg[64];
    #pragma unroll
    for (int j = 0; j < 16; ++j) {
      float4 f4 = ((const float4*)rp)[j];
      vreg[4*j+0] = f4.x; vreg[4*j+1] = f4.y; vreg[4*j+2] = f4.z; vreg[4*j+3] = f4.w;
    }
    float mx = -3.4e38f;
    #pragma unroll
    for (int j = 0; j < 64; ++j) mx = fmaxf(mx, vreg[j]);
    float sum = 0.f;
    #pragma unroll
    for (int j = 0; j < 64; ++j) { vreg[j] = __expf(vreg[j] - mx); sum += vreg[j]; }
    float inv = 1.0f / sum;
    // pack 4 probs per 8B store, row-private so in-place is safe
    unsigned short* pp = (unsigned short*)(sc + tid * 64);
    #pragma unroll
    for (int j = 0; j < 16; ++j) {
      union { unsigned short h[4]; uint2 u; } pk;
      pk.h[0] = f2bf(vreg[4*j+0] * inv); pk.h[1] = f2bf(vreg[4*j+1] * inv);
      pk.h[2] = f2bf(vreg[4*j+2] * inv); pk.h[3] = f2bf(vreg[4*j+3] * inv);
      *(uint2*)&pp[j * 4] = pk.u;
    }
  }
  __syncthreads();

  // ---- phase 3: out_h = P @ V  (32 tiles, K=64 -> 2 wmma steps) ----
  #pragma unroll 2
  for (int i = 0; i < 4; ++i) {
    int tt = w * 4 + i;
    int h = tt >> 3, qt = (tt >> 1) & 3, dt = tt & 1;
    const unsigned short* arow = (const unsigned short*)(sc + (h * 64 + qt * 16 + lm) * 64);
    const unsigned short* bcol = vs + (h * 32 + dt * 16 + lm) * 64;
    v8f acc = {};
    #pragma unroll
    for (int t = 0; t < 2; ++t) {
      v16bf a  = frag2(arow + t * 32 + lh * 8, arow + t * 32 + 16 + lh * 8);
      v16bf bb = frag2(bcol + t * 32 + lh * 8, bcol + t * 32 + 16 + lh * 8);
      acc = __builtin_amdgcn_wmma_f32_16x16x32_bf16(false, a, false, bb, (short)0, acc, false, false);
    }
    #pragma unroll
    for (int r = 0; r < 8; ++r) {
      int token = qt * 16 + r + 8 * lh;
      xs[token * XSTRIDE + h * 32 + dt * 16 + lm] = f2bf(acc[r]);  // x region recycled
    }
  }
  __syncthreads();

  // ---- phase 4: out = attn_out @ Wproj + b  (32 tiles, K=128 -> 4 wmma steps) ----
  float* outg = out + (size_t)blk * 8192;
  for (int i = 0; i < 4; ++i) {
    int tt = w * 4 + i;
    int mt = tt >> 3, ct = tt & 7;
    int col = ct * 16 + lm;
    const unsigned short* arow = xs + (mt * 16 + lm) * XSTRIDE;
    const unsigned short* bcol = pw_t + (size_t)col * 128;
    v8f acc = {};
    #pragma unroll
    for (int t = 0; t < 4; ++t) {
      v16bf a  = frag2(arow + t * 32 + lh * 8, arow + t * 32 + 16 + lh * 8);
      v16bf bb = frag2(bcol + t * 32 + lh * 8, bcol + t * 32 + 16 + lh * 8);
      acc = __builtin_amdgcn_wmma_f32_16x16x32_bf16(false, a, false, bb, (short)0, acc, false, false);
    }
    float bv = proj_b[col];
    #pragma unroll
    for (int r = 0; r < 8; ++r) {
      int token = mt * 16 + r + 8 * lh;
      outg[token * 128 + col] = acc[r] + bv;   // coalesced f32 stores
    }
  }
}

extern "C" void kernel_launch(void* const* d_in, const int* in_sizes, int n_in,
                              void* d_out, int out_size, void* d_ws, size_t ws_size,
                              hipStream_t stream) {
  const float* x      = (const float*)d_in[0];
  const float* qkv_w  = (const float*)d_in[1];
  const float* qkv_b  = (const float*)d_in[2];
  const float* proj_w = (const float*)d_in[3];
  const float* proj_b = (const float*)d_in[4];
  const float* btab   = (const float*)d_in[5];

  unsigned short* qw_t = (unsigned short*)d_ws;                     //  98304 B
  unsigned short* pw_t = (unsigned short*)((char*)d_ws + 98304);    //  32768 B
  float*          bias = (float*)((char*)d_ws + 131072);            //  65536 B

  swin_prep_kernel<<<320, 256, 0, stream>>>(qkv_w, proj_w, btab, qw_t, pw_t, bias);
  swin_main_kernel<<<4096, 256, 0, stream>>>(x, qkv_b, proj_b, qw_t, pw_t, bias,
                                             (float*)d_out);
}
